// KroneckerAttention_4526895530442
// MI455X (gfx1250) — compile-verified, matrix operand order checked by reading, not run on hardware
//
#include <hip/hip_runtime.h>
#include <hip/hip_bf16.h>
#include <math.h>

// ---------------- problem constants (B=2,H=16,N=4096,D=128, m=n=4) ----------
#define BHN   32      // b*h
#define NSEQ  4096
#define DHEAD 128
#define PGRP  1024    // p = q = N/4
#define KVT   32      // kv tile
#define MBLK  64      // query rows per block

typedef __attribute__((ext_vector_type(16))) __bf16 bf16x16;
typedef __attribute__((ext_vector_type(8)))  float  f32x8;

union BFrag { bf16x16 v; unsigned int u[8]; unsigned short s[16]; };

// pack two floats into one dword of bf16 (lo -> [15:0], hi -> [31:16]), RNE
__device__ __forceinline__ unsigned int f2bf2(float lo, float hi) {
  unsigned int ul = __builtin_bit_cast(unsigned int, lo);
  unsigned int uh = __builtin_bit_cast(unsigned int, hi);
  ul = (ul + 0x7FFFu + ((ul >> 16) & 1u)) >> 16;
  uh = (uh + 0x7FFFu + ((uh >> 16) & 1u)) & 0xFFFF0000u;
  return uh | ul;
}

// K-offset of packed pair v (0..7) for the 16-bit A/B fragment layout,
// per CDNA5 ISA 7.12.2 (16-bit A 16x32): v<4 -> 2v + 8*half ; v>=4 -> 16+2(v-4)+8*half
__device__ __forceinline__ int abOff(int v, int half) {
  return ((v & 4) ? (16 + ((v & 3) << 1)) : (v << 1)) + (half << 3);
}

// ---------------------------------------------------------------------------
// Kernel 1: partial sums of (q . k_sample)^2 per (bh, chunk-of-256-rows, g)
// ---------------------------------------------------------------------------
__global__ __launch_bounds__(256)
void knorm_kernel(const float* __restrict__ Q, const float* __restrict__ K,
                  const int* __restrict__ kp_ptr, float* __restrict__ partials) {
  const int chunk = blockIdx.x;      // 0..15  (256 rows each)
  const int bh    = blockIdx.y;      // 0..31
  const int tid   = threadIdx.x;     // 0..255
  const int kp    = *kp_ptr;

  __shared__ float ksamp[4][DHEAD];
  for (int i = tid; i < 4 * DHEAD; i += 256) {
    int g = i >> 7, d = i & 127;
    ksamp[g][d] = K[((size_t)bh * NSEQ + (size_t)g * PGRP + kp) * DHEAD + d];
  }
  __syncthreads();

  const int row = chunk * 256 + tid;
  const float* qr = Q + ((size_t)bh * NSEQ + row) * DHEAD;
  float dot[4] = {0.f, 0.f, 0.f, 0.f};
  for (int d = 0; d < DHEAD; d += 4) {
    float4 qv = *(const float4*)(qr + d);
#pragma unroll
    for (int g = 0; g < 4; ++g)
      dot[g] += qv.x * ksamp[g][d] + qv.y * ksamp[g][d + 1] +
                qv.z * ksamp[g][d + 2] + qv.w * ksamp[g][d + 3];
  }
  float sq[4];
#pragma unroll
  for (int g = 0; g < 4; ++g) sq[g] = dot[g] * dot[g];

  // wave32 reduce, then cross-wave via LDS (no atomics -> deterministic)
#pragma unroll
  for (int m = 1; m < 32; m <<= 1)
#pragma unroll
    for (int g = 0; g < 4; ++g) sq[g] += __shfl_xor(sq[g], m, 32);

  __shared__ float red[8][4];
  const int wid = tid >> 5, lane = tid & 31;
  if (lane == 0)
#pragma unroll
    for (int g = 0; g < 4; ++g) red[wid][g] = sq[g];
  __syncthreads();
  if (tid < 4) {
    float s = 0.f;
#pragma unroll
    for (int w = 0; w < 8; ++w) s += red[w][tid];
    partials[(size_t)bh * 64 + chunk * 4 + tid] = s;
  }
}

// ---------------------------------------------------------------------------
// Kernel 2: wts[bh][mi][g] and logS[bh][mi]
// ---------------------------------------------------------------------------
__global__ __launch_bounds__(128)
void kwts_kernel(const float* __restrict__ partials, float* __restrict__ wts,
                 float* __restrict__ logS) {
  const int t = threadIdx.x;
  if (t >= 128) return;
  const int bh = t >> 2, mi = t & 3;
  float nsq[4] = {0.f, 0.f, 0.f, 0.f};
#pragma unroll
  for (int cc = 0; cc < 4; ++cc)
#pragma unroll
    for (int g = 0; g < 4; ++g)
      nsq[g] += partials[(size_t)bh * 64 + (mi * 4 + cc) * 4 + g];
  float nrm[4], rs = 0.f;
#pragma unroll
  for (int g = 0; g < 4; ++g) { nrm[g] = sqrtf(nsq[g]); rs += nrm[g]; }
#pragma unroll
  for (int g = 0; g < 4; ++g) wts[(size_t)bh * 16 + mi * 4 + g] = nrm[g] / rs;
  float n22sq = 0.f;   // norms[c0=2][c1=2]^2
#pragma unroll
  for (int cc = 0; cc < 4; ++cc) n22sq += partials[(size_t)bh * 64 + (8 + cc) * 4 + 2];
  logS[(size_t)bh * 4 + mi] = logf(rs) - 0.5f * logf(n22sq);
}

// ---------------------------------------------------------------------------
// Kernel 3: WMMA flash attention (transposed: S^T = K Q^T, O^T = V^T P^T),
// double-buffered K/V tiles (compile-time buffer addresses), exp2 softmax,
// ballot-guarded accumulator rescale.
// ---------------------------------------------------------------------------
struct Tile  { unsigned short Ks[KVT][136]; unsigned short Vts[4][DHEAD][34]; };
struct EpiSh { float Og[4][16][132]; };
union  ShU   { Tile t[2]; EpiSh e; };

__device__ __forceinline__ f32x8 wmma_bf16(const bf16x16& a, const bf16x16& b, f32x8 c) {
  return __builtin_amdgcn_wmma_f32_16x16x32_bf16(false, a, false, b, (short)0, c,
                                                 false, false);
}

__global__ __launch_bounds__(512)
void kattn_kernel(const float* __restrict__ Q, const float* __restrict__ K,
                  const float* __restrict__ V, const float* __restrict__ wts,
                  const float* __restrict__ logS, float* __restrict__ outA,
                  float* __restrict__ outL) {
  const int qtile = blockIdx.x;              // 0..15 : 64-row tile of center group
  const int bh    = blockIdx.y;              // 0..31
  const int tid   = threadIdx.x;             // 0..511
  const int w     = tid >> 5;                // wave 0..15
  const int lane  = tid & 31;
  const int wq    = w >> 2;                  // row-tile 0..3 (16 rows each)
  const int wg    = w & 3;                   // value group 0..3
  const int half  = lane >> 4;
  const int r16   = lane & 15;               // A-row / B-col index of this lane

  __shared__ unsigned short Qs[MBLK][136];
  __shared__ ShU sh;
  __shared__ float wtsS[16], logSS[4];

  if (tid < 16) wtsS[tid] = wts[(size_t)bh * 16 + tid];
  if (tid < 4)  logSS[tid] = logS[(size_t)bh * 4 + tid];

  // fold 1/sqrt(128) * log2(e) into Q so softmax runs in exp2 domain
  const float scaleQ = 0.08838834764831845f * 1.4426950408889634f;
  const size_t baseQ = ((size_t)bh * NSEQ + 2 * PGRP + (size_t)qtile * MBLK) * DHEAD;
  const size_t baseK = ((size_t)bh * NSEQ + 2 * PGRP) * DHEAD;
  const size_t baseV = ((size_t)bh * NSEQ) * DHEAD;

  // ---- load Q tile (scaled, bf16, packed b64 stores) ----
  for (int i = tid; i < 2048; i += 512) {
    int r = i >> 5, c4 = (i & 31) << 2;
    float4 qv = *(const float4*)(Q + baseQ + (size_t)r * DHEAD + c4);
    uint2 pk;
    pk.x = f2bf2(qv.x * scaleQ, qv.y * scaleQ);
    pk.y = f2bf2(qv.z * scaleQ, qv.w * scaleQ);
    *(uint2*)&Qs[r][c4] = pk;
  }
  __syncthreads();

  // ---- preload Q^T B-fragments (this wave's 16 query rows) ----
  unsigned int qb[4][8];
  {
    const unsigned short* qrow = &Qs[wq * 16 + r16][0];
#pragma unroll
    for (int dc = 0; dc < 4; ++dc)
#pragma unroll
      for (int v = 0; v < 8; ++v)
        qb[dc][v] = *(const unsigned int*)(qrow + dc * 32 + abOff(v, half));
  }

  // ---- prefetch registers: K tile 2 float4, V tile 4x2 float4 (row pairs) ----
  float4 pK[2], pV0[4], pV1[4];
  auto loadRegs = [&](int kt) {
#pragma unroll
    for (int j = 0; j < 2; ++j) {
      int i = tid + j * 512, r = i >> 5, c4 = (i & 31) << 2;
      pK[j] = *(const float4*)(K + baseK + (size_t)(kt + r) * DHEAD + c4);
    }
#pragma unroll
    for (int j = 0; j < 4; ++j) {
      int i = tid + j * 512, g = i >> 9, rem = i & 511, rp = rem >> 5, c4 = (rem & 31) << 2;
      const float* rowb = V + baseV + (size_t)(g * PGRP + kt + 2 * rp) * DHEAD + c4;
      pV0[j] = *(const float4*)(rowb);
      pV1[j] = *(const float4*)(rowb + DHEAD);
    }
  };
  auto storeTile = [&](Tile& T) {
#pragma unroll
    for (int j = 0; j < 2; ++j) {
      int i = tid + j * 512, r = i >> 5, c4 = (i & 31) << 2;
      uint2 pk;
      pk.x = f2bf2(pK[j].x, pK[j].y);
      pk.y = f2bf2(pK[j].z, pK[j].w);
      *(uint2*)&T.Ks[r][c4] = pk;
    }
#pragma unroll
    for (int j = 0; j < 4; ++j) {
      int i = tid + j * 512, g = i >> 9, rem = i & 511, rp = rem >> 5, c4 = (rem & 31) << 2;
      float a0[4] = {pV0[j].x, pV0[j].y, pV0[j].z, pV0[j].w};
      float a1[4] = {pV1[j].x, pV1[j].y, pV1[j].z, pV1[j].w};
#pragma unroll
      for (int jj = 0; jj < 4; ++jj)  // pack two kv rows per b32 (kv contiguous in LDS)
        *(unsigned int*)&T.Vts[g][c4 + jj][2 * rp] = f2bf2(a0[jj], a1[jj]);
    }
  };

  f32x8 Oacc[8];
#pragma unroll
  for (int mt = 0; mt < 8; ++mt)
#pragma unroll
    for (int i = 0; i < 8; ++i) Oacc[mt][i] = 0.0f;
  float m_run = -3.0e38f, l_run = 0.0f;     // log2 domain

  // one KV step on tile T (compile-time LDS base), prefetching into Tn
  auto kvStep = [&](const Tile& T, Tile& Tn, int kt) {
    const bool hasNext = (kt + KVT) < PGRP;
    if (hasNext) loadRegs(kt + KVT);        // issue global loads; consumed after compute

    // ---- S^T = K_tile * Q^T : two 16x16 subtiles (kv 0..15 and 16..31) ----
    f32x8 s0, s1;
#pragma unroll
    for (int i = 0; i < 8; ++i) { s0[i] = 0.0f; s1[i] = 0.0f; }
#pragma unroll
    for (int dc = 0; dc < 4; ++dc) {
      BFrag a0, a1, b;
#pragma unroll
      for (int v = 0; v < 8; ++v) {
        const int off = dc * 32 + abOff(v, half);
        a0.u[v] = *(const unsigned int*)&T.Ks[r16][off];
        a1.u[v] = *(const unsigned int*)&T.Ks[16 + r16][off];
        b.u[v]  = qb[dc][v];
      }
      s0 = wmma_bf16(a0.v, b.v, s0);
      s1 = wmma_bf16(a1.v, b.v, s1);
    }

    // ---- online softmax, exp2 domain (each lane owns query column r16) ----
    float mloc = s0[0];
#pragma unroll
    for (int i = 0; i < 8; ++i) { mloc = fmaxf(mloc, s0[i]); mloc = fmaxf(mloc, s1[i]); }
    mloc = fmaxf(mloc, __shfl_xor(mloc, 16, 32));      // merge half-pair (full 32 kv)
    const float mnew = fmaxf(m_run, mloc);
    if (__ballot(mloc > m_run)) {           // wave-uniform: rescale only on new max
      const float corr = exp2f(m_run - mnew);
      l_run *= corr;
#pragma unroll
      for (int mt = 0; mt < 8; ++mt)
#pragma unroll
        for (int i = 0; i < 8; ++i) Oacc[mt][i] *= corr;
    }
    m_run = mnew;
    float p0[8], p1[8], lloc = 0.0f;
#pragma unroll
    for (int i = 0; i < 8; ++i) {
      p0[i] = exp2f(s0[i] - mnew); lloc += p0[i];
      p1[i] = exp2f(s1[i] - mnew); lloc += p1[i];
    }
    lloc += __shfl_xor(lloc, 16, 32);
    l_run += lloc;

    // P^T already in B-fragment layout: slot e<8 -> kv 8*half+e, e>=8 -> 16+8*half+(e-8)
    BFrag pb;
#pragma unroll
    for (int v = 0; v < 4; ++v) {
      pb.u[v]     = f2bf2(p0[2 * v], p0[2 * v + 1]);
      pb.u[4 + v] = f2bf2(p1[2 * v], p1[2 * v + 1]);
    }

    // ---- O^T += V^T * P^T : 8 value-column tiles of 16 ----
#pragma unroll
    for (int mt = 0; mt < 8; ++mt) {
      BFrag a;
#pragma unroll
      for (int v = 0; v < 8; ++v)
        a.u[v] = *(const unsigned int*)&T.Vts[wg][mt * 16 + r16][abOff(v, half)];
      Oacc[mt] = wmma_bf16(a.v, pb.v, Oacc[mt]);
    }

    if (hasNext) storeTile(Tn);             // convert + ds_store next tile
    __syncthreads();
  };

  loadRegs(0);
  storeTile(sh.t[0]);
  __syncthreads();

  // =========== KV loop, unrolled x2 so LDS buffer offsets are immediates ====
  for (int kt = 0; kt < PGRP; kt += 2 * KVT) {
    kvStep(sh.t[0], sh.t[1], kt);
    kvStep(sh.t[1], sh.t[0], kt + KVT);
  }

  // =============================== epilogue ===============================
  const float inv_l = 1.0f / l_run;
  const float lse0  = (m_run + log2f(l_run)) * 0.6931471805599453f;  // back to ln
#pragma unroll
  for (int mt = 0; mt < 8; ++mt)
#pragma unroll
    for (int i = 0; i < 8; ++i) Oacc[mt][i] *= inv_l;

  if (wg == 0 && half == 0) {   // lse for this wave's 16 query rows, all 4 mi blocks
    const int qi = qtile * MBLK + wq * 16 + r16;
#pragma unroll
    for (int mi = 0; mi < 4; ++mi)
      outL[(size_t)bh * NSEQ + (size_t)mi * PGRP + qi] = lse0 + logSS[mi];
  }

  // stage O^T per row-tile through LDS and apply Kronecker weights
  for (int wqs = 0; wqs < 4; ++wqs) {
    __syncthreads();
    if (wq == wqs) {
#pragma unroll
      for (int mt = 0; mt < 8; ++mt)
#pragma unroll
        for (int vg = 0; vg < 8; ++vg)
          sh.e.Og[wg][r16][mt * 16 + (half << 3) + vg] = Oacc[mt][vg];
    }
    __syncthreads();
    for (int e = tid; e < 4 * 16 * DHEAD; e += 512) {
      const int mi = e >> 11, rem = e & 2047, qi = rem >> 7, col = rem & 127;
      float acc = 0.0f;
#pragma unroll
      for (int g = 0; g < 4; ++g) acc += wtsS[mi * 4 + g] * sh.e.Og[g][qi][col];
      const size_t row = (size_t)bh * NSEQ + (size_t)mi * PGRP + qtile * MBLK + wqs * 16 + qi;
      outA[row * DHEAD + col] = acc;
    }
  }
}

// ---------------------------------------------------------------------------
extern "C" void kernel_launch(void* const* d_in, const int* in_sizes, int n_in,
                              void* d_out, int out_size, void* d_ws, size_t ws_size,
                              hipStream_t stream) {
  (void)in_sizes; (void)n_in; (void)out_size; (void)ws_size;
  const float* Q = (const float*)d_in[0];
  const float* K = (const float*)d_in[1];
  const float* V = (const float*)d_in[2];
  const int* kp  = (const int*)d_in[5];

  float* ws       = (float*)d_ws;
  float* partials = ws;            // 32*16*4  = 2048 floats
  float* wts      = ws + 2048;     // 32*4*4   = 512 floats
  float* logS     = ws + 2560;     // 32*4     = 128 floats

  float* outA = (float*)d_out;                         // (2,16,4096,128)
  float* outL = outA + (size_t)BHN * NSEQ * DHEAD;     // (2,16,4096,1)

  knorm_kernel<<<dim3(16, BHN), 256, 0, stream>>>(Q, K, kp, partials);
  kwts_kernel<<<1, 128, 0, stream>>>(partials, wts, logS);
  kattn_kernel<<<dim3(16, BHN), 512, 0, stream>>>(Q, K, V, wts, logS, outA, outL);
}
